// SCC_31988916420818
// MI455X (gfx1250) — matrix operand order repaired and checked by reading, not compiled, not run on hardware
//
#include <hip/hip_runtime.h>
#include <hip/hip_bf16.h>
#include <stdint.h>

// ---------------------------------------------------------------------------
// CDNA5 (gfx1250) wave32 WMMA types and fragment loaders
// ---------------------------------------------------------------------------
typedef __attribute__((ext_vector_type(16))) __bf16 v16bf;
typedef __attribute__((ext_vector_type(8)))  float  v8f;
typedef __attribute__((ext_vector_type(2)))  float  v2f;

#ifndef __has_builtin
#define __has_builtin(x) 0
#endif

// A fragment, 16x32 bf16, logical A[m][k] row-major with stride lda (even),
// k0 multiple of 32.  ISA layout: lanes 0-15: M=lane, K pairs {0..7,16..23};
// lanes 16-31: M=lane-16, K pairs {8..15,24..31}.
__device__ __forceinline__ v16bf frag_a_bf16(const __bf16* A, int lda, int m0, int k0) {
  const int lane = threadIdx.x & 31;
  const int m = m0 + (lane & 15);
  const int kb = (lane >> 4) << 3;                     // 0 or 8
  const uint32_t* p = (const uint32_t*)(A + (size_t)lda * m + k0 + kb);
  v16bf r; uint32_t* r32 = (uint32_t*)&r;
  r32[0] = p[0]; r32[1] = p[1]; r32[2] = p[2]; r32[3] = p[3];
  r32[4] = p[8]; r32[5] = p[9]; r32[6] = p[10]; r32[7] = p[11];
  return r;
}

// B fragment, 32x16 bf16, from TRANSPOSED storage BT[n][k] row-major,
// stride ldb (even), k0 multiple of 32.  ISA layout: lanes 0-15: N=lane,
// K {0..7,16..23}; lanes 16-31: N=lane-16, K {8..15,24..31}.
__device__ __forceinline__ v16bf frag_b_bf16(const __bf16* BT, int ldb, int n0, int k0) {
  const int lane = threadIdx.x & 31;
  const int n = n0 + (lane & 15);
  const int kh = (lane >> 4) << 3;                     // 0 or 8
  const uint32_t* p = (const uint32_t*)(BT + (size_t)ldb * n + k0 + kh);
  v16bf r; uint32_t* r32 = (uint32_t*)&r;
#pragma unroll
  for (int j = 0; j < 8; ++j) r32[j] = p[j < 4 ? j : j + 4];
  return r;
}

// A fragment from column-major storage (logical A[m][k] stored Acm[k*lda+m]).
__device__ __forceinline__ v16bf frag_a_bf16_tr(const __bf16* Acm, int lda, int m0, int k0) {
  const int lane = threadIdx.x & 31;
  const int m = m0 + (lane & 15);
  const int kb = (lane >> 4) << 3;
  v16bf r;
#pragma unroll
  for (int e = 0; e < 16; ++e) {
    int j = e >> 1;
    int k = k0 + ((j >= 4) ? 16 : 0) + kb + ((j & 3) << 1) + (e & 1);
    r[e] = Acm[(size_t)k * lda + m];
  }
  return r;
}

// B fragment from NON-transposed storage B[k][n] row-major (gather).
__device__ __forceinline__ v16bf frag_b_bf16_nt(const __bf16* B, int ldb, int n0, int k0) {
  const int lane = threadIdx.x & 31;
  const int n = n0 + (lane & 15);
  const int kh = (lane >> 4) << 3;
  v16bf r;
#pragma unroll
  for (int e = 0; e < 16; ++e) r[e] = B[(size_t)(k0 + kh + e + ((e >> 3) << 3)) * ldb + n];
  return r;
}

// f32 WMMA 16x16x4 fragments: A 16x4, lanes 0-15 M=lane K={0,1}, lanes 16-31 K={2,3}
__device__ __forceinline__ v2f frag_a_f32(const float* A, int lda, int m0, int k0) {
  const int lane = threadIdx.x & 31;
  const int m = m0 + (lane & 15);
  const int ks = (lane >> 4) << 1;
  v2f r; r.x = A[(size_t)lda * m + k0 + ks]; r.y = A[(size_t)lda * m + k0 + ks + 1];
  return r;
}
__device__ __forceinline__ v2f frag_b_f32(const float* BT, int ldb, int n0, int k0) {
  const int lane = threadIdx.x & 31;
  const int n = n0 + (lane & 15);
  const int ks = (lane >> 4) << 1;
  v2f r; r.x = BT[(size_t)ldb * n + k0 + ks]; r.y = BT[(size_t)ldb * n + k0 + ks + 1];
  return r;
}

#define WMMA_BF16(a, b, c) \
  __builtin_amdgcn_wmma_f32_16x16x32_bf16(false, (a), false, (b), (short)0, (c), false, false)
#define WMMA_F32X4(a, b, c) \
  __builtin_amdgcn_wmma_f32_16x16x4_f32(false, (a), false, (b), (short)0, (c), false, false)

__device__ __forceinline__ float lrelu(float v) { return v >= 0.f ? v : 0.2f * v; }

// ---------------------------------------------------------------------------
// Probe kernels (compiled, never launched): CDNA5 async/cluster/TDM paths
// ---------------------------------------------------------------------------
__global__ void k_cdna5_probe(const float* __restrict__ g, float* __restrict__ o) {
  __shared__ float buf[256];
  const int tid = threadIdx.x;
  unsigned lds_byte = tid * 4u;
  const float* gp = g + tid;
  asm volatile("global_load_async_to_lds_b32 %0, %1, off"
               :
               : "v"(lds_byte), "v"(gp)
               : "memory");
  asm volatile("s_wait_asynccnt 0" ::: "memory");
  __syncthreads();
  float v = buf[tid];
  v += (float)__builtin_amdgcn_cluster_id_x();
  __builtin_amdgcn_s_cluster_barrier();
  o[tid] = v;
}

#if __has_builtin(__builtin_amdgcn_tensor_load_to_lds)
typedef __attribute__((ext_vector_type(4))) unsigned tdm_u32x4;
typedef __attribute__((ext_vector_type(8))) int      tdm_i32x8;
typedef __attribute__((ext_vector_type(4))) int      tdm_i32x4;
__global__ void k_tdm_probe() {
  // amdgpu-toolchain (clang-23) 6-arg form:
  // (u32x4 g0, i32x8 g1, i32x4 g2, i32x4 g3, i32x8 g4, i32 cpol)
  tdm_u32x4 g0 = {};
  tdm_i32x8 g1 = {};
  tdm_i32x4 g2 = {};
  tdm_i32x4 g3 = {};
  tdm_i32x8 g4 = {};
  __builtin_amdgcn_tensor_load_to_lds(g0, g1, g2, g3, g4, 0);
  __builtin_amdgcn_s_wait_tensorcnt(0);
}
#endif

// ---------------------------------------------------------------------------
// K1: t1 = lrelu(x(NCHW) . W1 + b1), M=262144 pixels, N=36(pad48), K=180(pad192)
// ---------------------------------------------------------------------------
__global__ __launch_bounds__(256) void k_dfe1(const float* __restrict__ x,
                                              const float* __restrict__ w1,
                                              const float* __restrict__ b1,
                                              __bf16* __restrict__ t1) {
  __shared__ __bf16 sA[64 * 192];
  __shared__ __bf16 sBT[48 * 192];
  const int tid = threadIdx.x;
  const long Pbase = (long)blockIdx.x * 64;
  const long b = Pbase >> 16;
  const long hw0 = Pbase & 65535;
  for (int idx = tid; idx < 180 * 16; idx += 256) {          // float4 staging
    int k = idx >> 4, i4 = idx & 15;
    float4 v = *(const float4*)(x + ((b * 180 + k) << 16) + hw0 + i4 * 4);
    int i = i4 * 4;
    sA[(i + 0) * 192 + k] = (__bf16)v.x;
    sA[(i + 1) * 192 + k] = (__bf16)v.y;
    sA[(i + 2) * 192 + k] = (__bf16)v.z;
    sA[(i + 3) * 192 + k] = (__bf16)v.w;
  }
  for (int idx = tid; idx < 64 * 12; idx += 256) {           // K padding
    int i = idx / 12, k = 180 + idx % 12;
    sA[i * 192 + k] = (__bf16)0.f;
  }
  for (int idx = tid; idx < 48 * 192; idx += 256) {
    int o = idx / 192, c = idx - o * 192;
    float v = (o < 36 && c < 180) ? w1[c * 36 + o] : 0.f;
    sBT[o * 192 + c] = (__bf16)v;
  }
  __syncthreads();
  const int wave = tid >> 5, lane = tid & 31;
  for (int t = wave; t < 12; t += 8) {
    int msub = t & 3, nt = t >> 2;
    v8f acc = {};
#pragma unroll
    for (int kc = 0; kc < 6; ++kc) {
      v16bf a = frag_a_bf16(sA, 192, msub * 16, kc * 32);
      v16bf bb = frag_b_bf16(sBT, 192, nt * 16, kc * 32);
      acc = WMMA_BF16(a, bb, acc);
    }
    int n = nt * 16 + (lane & 15);
    if (n < 36) {
      float bias = b1[n];
#pragma unroll
      for (int r = 0; r < 8; ++r) {
        int m = r + ((lane >> 4) << 3);
        long P = Pbase + msub * 16 + m;
        t1[P * 36 + n] = (__bf16)lrelu(acc[r] + bias);
      }
    }
  }
}

// ---------------------------------------------------------------------------
// K2: t2 = lrelu(conv3x3(t1)), im2col GEMM K=324(pad352), N=36(pad48)
// ---------------------------------------------------------------------------
__global__ __launch_bounds__(256) void k_dfe2(const __bf16* __restrict__ t1,
                                              const float* __restrict__ w2,
                                              const float* __restrict__ b2,
                                              __bf16* __restrict__ t2) {
  __shared__ __bf16 halo[3 * 66 * 36];
  __shared__ __bf16 sA[64 * 32];
  __shared__ __bf16 sBT[48 * 32];
  const int tid = threadIdx.x;
  const long Pbase = (long)blockIdx.x * 64;
  const long b = Pbase >> 16;
  const int h = (int)((Pbase >> 8) & 255);
  const int w0 = (int)(Pbase & 255);
  for (int idx = tid; idx < 3 * 66 * 36; idx += 256) {
    int c = idx % 36; int t = idx / 36; int cx = t % 66; int ry = t / 66;
    int hh = h + ry - 1, wc = w0 + cx - 1;
    __bf16 v = (__bf16)0.f;
    if (hh >= 0 && hh < 256 && wc >= 0 && wc < 256) {
      long P = (b << 16) + ((long)hh << 8) + wc;
      v = t1[P * 36 + c];
    }
    halo[idx] = v;
  }
  __syncthreads();
  const int wave = tid >> 5, lane = tid & 31;
  v8f acc[2] = {{}, {}};
  for (int kc = 0; kc < 11; ++kc) {
    __syncthreads();
    for (int idx = tid; idx < 64 * 32; idx += 256) {
      int i = idx >> 5, kk = idx & 31;
      int k = kc * 32 + kk;
      __bf16 v = (__bf16)0.f;
      if (k < 324) {
        int dy = k / 108; int rr = k - dy * 108; int dx = rr / 36; int c = rr - dx * 36;
        v = halo[((dy * 66) + (i + dx)) * 36 + c];
      }
      sA[i * 32 + kk] = v;
    }
    for (int idx = tid; idx < 48 * 32; idx += 256) {
      int o = idx >> 5, kk = idx & 31;
      int k = kc * 32 + kk;
      float v = (o < 36 && k < 324) ? w2[k * 36 + o] : 0.f;
      sBT[o * 32 + kk] = (__bf16)v;
    }
    __syncthreads();
#pragma unroll
    for (int s = 0; s < 2; ++s) {
      int t = wave + 8 * s;
      if (t < 12) {
        int msub = t & 3, nt = t >> 2;
        v16bf a = frag_a_bf16(sA, 32, msub * 16, 0);
        v16bf bb = frag_b_bf16(sBT, 32, nt * 16, 0);
        acc[s] = WMMA_BF16(a, bb, acc[s]);
      }
    }
  }
#pragma unroll
  for (int s = 0; s < 2; ++s) {
    int t = wave + 8 * s;
    if (t >= 12) continue;
    int msub = t & 3, nt = t >> 2;
    int n = nt * 16 + (lane & 15);
    if (n < 36) {
      float bias = b2[n];
#pragma unroll
      for (int r = 0; r < 8; ++r) {
        int m = r + ((lane >> 4) << 3);
        long P = Pbase + msub * 16 + m;
        t2[P * 36 + n] = (__bf16)lrelu(acc[s][r] + bias);
      }
    }
  }
}

// ---------------------------------------------------------------------------
// K3: qv = (t2.W3 + b3) * (x.Wl + bl), 64-pixel tiles, N=180(pad192)
// ---------------------------------------------------------------------------
__global__ __launch_bounds__(256) void k_gate(const float* __restrict__ x,
                                              const __bf16* __restrict__ t2,
                                              const float* __restrict__ w3,
                                              const float* __restrict__ b3,
                                              const float* __restrict__ wl,
                                              const float* __restrict__ bl,
                                              float* __restrict__ qv) {
  __shared__ __bf16 sA1[64 * 64];
  __shared__ __bf16 sA2[64 * 192];
  __shared__ __bf16 sBT[192 * 32];
  const int tid = threadIdx.x;
  const long Pbase = (long)blockIdx.x * 64;
  const long b = Pbase >> 16;
  const long hw0 = Pbase & 65535;
  for (int idx = tid; idx < 64 * 64; idx += 256) {
    int i = idx >> 6, k = idx & 63;
    sA1[i * 64 + k] = (k < 36) ? t2[(Pbase + i) * 36 + k] : (__bf16)0.f;
  }
  for (int idx = tid; idx < 180 * 16; idx += 256) {          // float4 staging of x
    int k = idx >> 4, i4 = idx & 15;
    float4 v = *(const float4*)(x + ((b * 180 + k) << 16) + hw0 + i4 * 4);
    int i = i4 * 4;
    sA2[(i + 0) * 192 + k] = (__bf16)v.x;
    sA2[(i + 1) * 192 + k] = (__bf16)v.y;
    sA2[(i + 2) * 192 + k] = (__bf16)v.z;
    sA2[(i + 3) * 192 + k] = (__bf16)v.w;
  }
  for (int idx = tid; idx < 64 * 12; idx += 256) {
    int i = idx / 12, k = 180 + idx % 12;
    sA2[i * 192 + k] = (__bf16)0.f;
  }
  const int wave = tid >> 5, lane = tid & 31;
  v8f acc1[6] = {{}, {}, {}, {}, {}, {}};
  v8f acc2[6] = {{}, {}, {}, {}, {}, {}};
  for (int kc = 0; kc < 2; ++kc) {           // t2 . W3  (K=36)
    __syncthreads();
    for (int idx = tid; idx < 192 * 32; idx += 256) {
      int o = idx >> 5, kk = idx & 31;
      int k = kc * 32 + kk;
      float v = (o < 180 && k < 36) ? w3[k * 180 + o] : 0.f;
      sBT[o * 32 + kk] = (__bf16)v;
    }
    __syncthreads();
#pragma unroll
    for (int s = 0; s < 6; ++s) {
      int t = wave + 8 * s;
      int msub = t & 3, nt = t >> 2;
      v16bf a = frag_a_bf16(sA1, 64, msub * 16, kc * 32);
      v16bf bb = frag_b_bf16(sBT, 32, nt * 16, 0);
      acc1[s] = WMMA_BF16(a, bb, acc1[s]);
    }
  }
  for (int kc = 0; kc < 6; ++kc) {           // x . Wl (K=180)
    __syncthreads();
    for (int idx = tid; idx < 192 * 32; idx += 256) {
      int o = idx >> 5, kk = idx & 31;
      int k = kc * 32 + kk;
      float v = (o < 180 && k < 180) ? wl[k * 180 + o] : 0.f;
      sBT[o * 32 + kk] = (__bf16)v;
    }
    __syncthreads();
#pragma unroll
    for (int s = 0; s < 6; ++s) {
      int t = wave + 8 * s;
      int msub = t & 3, nt = t >> 2;
      v16bf a = frag_a_bf16(sA2, 192, msub * 16, kc * 32);
      v16bf bb = frag_b_bf16(sBT, 32, nt * 16, 0);
      acc2[s] = WMMA_BF16(a, bb, acc2[s]);
    }
  }
#pragma unroll
  for (int s = 0; s < 6; ++s) {
    int t = wave + 8 * s;
    int msub = t & 3, nt = t >> 2;
    int n = nt * 16 + (lane & 15);
    if (n < 180) {
      float bb3 = b3[n], bbl = bl[n];
#pragma unroll
      for (int r = 0; r < 8; ++r) {
        int m = r + ((lane >> 4) << 3);
        qv[(Pbase + msub * 16 + m) * 180 + n] = (acc1[s][r] + bb3) * (acc2[s][r] + bbl);
      }
    }
  }
}

// ---------------------------------------------------------------------------
// K4: positional-bias MLP + rpb reduction (tiny, scalar)
// ---------------------------------------------------------------------------
__global__ __launch_bounds__(256) void k_pos(
    const float* __restrict__ pw, const float* __restrict__ pb,
    const float* __restrict__ ln1w, const float* __restrict__ ln1b,
    const float* __restrict__ l1w, const float* __restrict__ l1b,
    const float* __restrict__ ln2w, const float* __restrict__ ln2b,
    const float* __restrict__ l2w, const float* __restrict__ l2b,
    const float* __restrict__ ln3w, const float* __restrict__ ln3b,
    const float* __restrict__ l3w, const float* __restrict__ l3b,
    float* __restrict__ rpb) {
  __shared__ float pos[961 * 6];
  const int tid = threadIdx.x;
  for (int a = tid; a < 961; a += 256) {
    float cur[11], nxt[11];
    float bh = (float)(a / 31) - 15.f;
    float bw = (float)(a % 31) - 15.f;
    for (int j = 0; j < 11; ++j) cur[j] = bh * pw[j] + bw * pw[11 + j] + pb[j];
    auto stage = [&](const float* lw_, const float* lb_, const float* W,
                     const float* bias, int nout, float* out) {
      float m = 0.f;
      for (int j = 0; j < 11; ++j) m += cur[j];
      m *= (1.f / 11.f);
      float var = 0.f;
      for (int j = 0; j < 11; ++j) { float d = cur[j] - m; var += d * d; }
      var *= (1.f / 11.f);
      float inv = rsqrtf(var + 1e-5f);
      float t[11];
      for (int j = 0; j < 11; ++j) {
        float u = (cur[j] - m) * inv * lw_[j] + lb_[j];
        t[j] = u > 0.f ? u : 0.f;
      }
      for (int o = 0; o < nout; ++o) {
        float s = bias[o];
        for (int j = 0; j < 11; ++j) s += t[j] * W[j * nout + o];
        out[o] = s;
      }
    };
    stage(ln1w, ln1b, l1w, l1b, 11, nxt);
    for (int j = 0; j < 11; ++j) cur[j] = nxt[j];
    stage(ln2w, ln2b, l2w, l2b, 11, nxt);
    for (int j = 0; j < 11; ++j) cur[j] = nxt[j];
    stage(ln3w, ln3b, l3w, l3b, 6, nxt);
    for (int h2 = 0; h2 < 6; ++h2) pos[a * 6 + h2] = nxt[h2];
  }
  __syncthreads();
  for (int idx = tid; idx < 6 * 256 * 64; idx += 256) {
    int h2 = idx >> 14;
    int rem = idx & 16383;
    int l = rem >> 6, m = rem & 63;
    int lh = l >> 4, lw2 = l & 15;
    int mh = m >> 3, mw = m & 7;
    float s = 0.f;
    for (int i = 0; i < 2; ++i)
      for (int j = 0; j < 2; ++j) {
        int r0 = lh - (2 * mh + i) + 15;
        int r1 = lw2 - (2 * mw + j) + 15;
        s += pos[(r0 * 31 + r1) * 6 + h2];
      }
    rpb[idx] = 0.25f * s;
  }
}

// ---------------------------------------------------------------------------
// K5: per-window spatial attention (f32 WMMA 16x16x4, full fp32 precision)
// ---------------------------------------------------------------------------
__global__ __launch_bounds__(256) void k_attn_sp(const float* __restrict__ qv,
                                                 const float* __restrict__ slw,
                                                 const float* __restrict__ slb,
                                                 const float* __restrict__ rpb,
                                                 float* __restrict__ y) {
  __shared__ __align__(16) unsigned char smem[65536];
  float* vf = (float*)smem;                       // 256*16
  float* qf = (float*)(smem + 16384);             // 128*16
  float* vpB = (float*)(smem + 24576);            // 64*16   (BT for GEMM1)
  float* vpT = (float*)(smem + 28672);            // 16*64   (BT for GEMM2)
  float* corr = (float*)(smem + 32768);           // 128*64
  const int tid = threadIdx.x;
  const int wv = blockIdx.x;
  const long b = wv >> 8;
  const int wy = (wv >> 4) & 15, wx = wv & 15;
  auto gpix = [&](int l) -> long {
    int ly = l >> 4, lx = l & 15;
    return (b << 16) + ((long)((wy << 4) + ly) << 8) + ((wx << 4) + lx);
  };
  const float s0 = slw[0], s1 = slw[1], s2 = slw[2], s3 = slw[3], sb = slb[0];
  for (int head = 0; head < 6; ++head) {
    __syncthreads();
    for (int idx = tid; idx < 256 * 16; idx += 256) {
      int l = idx >> 4, d = idx & 15;
      float vvv = 0.f;
      if (d < 15) vvv = qv[gpix(l) * 180 + 90 + head * 15 + d];
      vf[idx] = vvv;
    }
    __builtin_prefetch(&qv[gpix(0) * 180 + head * 15], 0, 1);
    __syncthreads();
    for (int idx = tid; idx < 64 * 16; idx += 256) {
      int m = idx >> 4, d = idx & 15;
      float v = 0.f;
      if (d < 15) {
        int mh = m >> 3, mw = m & 7;
        int l00 = ((mh * 2) << 4) + mw * 2;
        v = sb + s0 * vf[l00 * 16 + d] + s1 * vf[(l00 + 1) * 16 + d] +
            s2 * vf[(l00 + 16) * 16 + d] + s3 * vf[(l00 + 17) * 16 + d];
      }
      vpB[m * 16 + d] = v;
      vpT[d * 64 + m] = v;
    }
    for (int half = 0; half < 2; ++half) {
      __syncthreads();
      for (int idx = tid; idx < 128 * 16; idx += 256) {
        int l = idx >> 4, d = idx & 15;
        float qq = 0.f;
        if (d < 15) qq = qv[gpix(half * 128 + l) * 180 + head * 15 + d];
        qf[idx] = qq;
      }
      __syncthreads();
      const int wave = tid >> 5, lane = tid & 31;
      for (int t = wave; t < 32; t += 8) {           // corr = q . vp^T / 15 + rpb
        int msub = t >> 2, nt = t & 3;
        v8f acc = {};
#pragma unroll
        for (int kc = 0; kc < 4; ++kc) {
          v2f a = frag_a_f32(qf, 16, msub * 16, kc * 4);
          v2f bb = frag_b_f32(vpB, 16, nt * 16, kc * 4);
          acc = WMMA_F32X4(a, bb, acc);
        }
        int mcol = nt * 16 + (lane & 15);
#pragma unroll
        for (int r = 0; r < 8; ++r) {
          int ll = msub * 16 + r + ((lane >> 4) << 3);
          int gl = half * 128 + ll;
          corr[ll * 64 + mcol] =
              acc[r] * (1.f / 15.f) + rpb[(head * 256 + gl) * 64 + mcol];
        }
      }
      __syncthreads();
      {                                             // xs = corr . vp
        int msub = wave;
        v8f acc = {};
#pragma unroll
        for (int kc = 0; kc < 16; ++kc) {
          v2f a = frag_a_f32(corr, 64, msub * 16, kc * 4);
          v2f bb = frag_b_f32(vpT, 64, 0, kc * 4);
          acc = WMMA_F32X4(a, bb, acc);
        }
        int d = lane & 15;
        if (d < 15) {
#pragma unroll
          for (int r = 0; r < 8; ++r) {
            int ll = msub * 16 + r + ((lane >> 4) << 3);
            y[gpix(half * 128 + ll) * 180 + head * 15 + d] = acc[r];
          }
        }
      }
      __syncthreads();
    }
  }
}

// ---------------------------------------------------------------------------
// K6: per-window channel correlation (bf16 WMMA)
// ---------------------------------------------------------------------------
__global__ __launch_bounds__(256) void k_attn_ch(const float* __restrict__ qv,
                                                 float* __restrict__ y) {
  __shared__ __align__(16) unsigned char smem[49152];
  const int tid = threadIdx.x;
  const int wave = tid >> 5, lane = tid & 31;
  const int wv = blockIdx.x;
  const long b = wv >> 8;
  const int wy = (wv >> 4) & 15, wx = wv & 15;
  auto gpix = [&](int l) -> long {
    int ly = l >> 4, lx = l & 15;
    return (b << 16) + ((long)((wy << 4) + ly) << 8) + ((wx << 4) + lx);
  };
  __bf16* qcN = (__bf16*)smem;             // 128*96, qc[l][c]
  __bf16* vcN = (__bf16*)(smem + 24576);   // 128*96, vc[l][c]
  v8f acc[5] = {{}, {}, {}, {}, {}};
  for (int lp = 0; lp < 2; ++lp) {         // cmap = qc^T . vc / 256, K split over l
    __syncthreads();
    for (int idx = tid; idx < 128 * 96; idx += 256) {
      int l = idx / 96, c = idx - l * 96;
      int gl = lp * 128 + l;
      float qq = 0.f, vvv = 0.f;
      if (c < 90) {
        long base = gpix(gl) * 180 + c;
        qq = qv[base];
        vvv = qv[base + 90];
      }
      qcN[l * 96 + c] = (__bf16)qq;
      vcN[l * 96 + c] = (__bf16)vvv;
    }
    __syncthreads();
    for (int s = 0; s < 5; ++s) {
      int t = wave + 8 * s;
      if (t >= 36) break;
      int mc = t / 6, nd = t - mc * 6;
#pragma unroll
      for (int kc = 0; kc < 4; ++kc) {
        v16bf a = frag_a_bf16_tr(qcN, 96, mc * 16, kc * 32);   // A[c][l] = qc[l][c]
        v16bf bb = frag_b_bf16_nt(vcN, 96, nd * 16, kc * 32);  // B[l][d] = vc[l][d]
        acc[s] = WMMA_BF16(a, bb, acc[s]);
      }
    }
  }
  __syncthreads();
  __bf16* cmapT = (__bf16*)smem;           // 96*96, cmap[c][d] (overwrites qcN)
  for (int s = 0; s < 5; ++s) {
    int t = wave + 8 * s;
    if (t >= 36) break;
    int mc = t / 6, nd = t - mc * 6;
    int d = nd * 16 + (lane & 15);
#pragma unroll
    for (int r = 0; r < 8; ++r) {
      int c = mc * 16 + r + ((lane >> 4) << 3);
      cmapT[c * 96 + d] = (__bf16)(acc[s][r] * (1.f / 256.f));
    }
  }
  __syncthreads();
  __bf16* vc2 = (__bf16*)(smem + 18432);   // 128*96
  for (int lp = 0; lp < 2; ++lp) {         // xc = vc . cmap^T
    __syncthreads();
    for (int idx = tid; idx < 128 * 96; idx += 256) {
      int l = idx / 96, c = idx - l * 96;
      float vvv = 0.f;
      if (c < 90) vvv = qv[gpix(lp * 128 + l) * 180 + 90 + c];
      vc2[l * 96 + c] = (__bf16)vvv;
    }
    __syncthreads();
    for (int t = wave; t < 48; t += 8) {
      int ml = t / 6, nc = t - ml * 6;
      v8f a2 = {};
#pragma unroll
      for (int kc = 0; kc < 3; ++kc) {
        v16bf a = frag_a_bf16(vc2, 96, ml * 16, kc * 32);
        v16bf bb = frag_b_bf16(cmapT, 96, nc * 16, kc * 32);  // BT[n=c][k=d]
        a2 = WMMA_BF16(a, bb, a2);
      }
      int c = nc * 16 + (lane & 15);
      if (c < 90) {
#pragma unroll
        for (int r = 0; r < 8; ++r) {
          int l = lp * 128 + ml * 16 + r + ((lane >> 4) << 3);
          y[gpix(l) * 180 + 90 + c] = a2[r];
        }
      }
    }
    __syncthreads();
  }
}

// ---------------------------------------------------------------------------
// K7: out = x + rmsnorm(y . Wp + bp) * norm_w, 64-pixel tiles (~88KB LDS probe)
// ---------------------------------------------------------------------------
__global__ __launch_bounds__(256) void k_proj(const float* __restrict__ x,
                                              const float* __restrict__ y,
                                              const float* __restrict__ pw,
                                              const float* __restrict__ pb,
                                              const float* __restrict__ nw,
                                              float* __restrict__ out) {
  __shared__ __bf16 sA[64 * 192];    // 24.6 KB
  __shared__ __bf16 sBT[192 * 32];   // 12.3 KB
  __shared__ float sOut[64 * 192];   // 49.2 KB
  __shared__ float sPart[64 * 4];
  __shared__ float sScale[64];
  const int tid = threadIdx.x;
  const long Pbase = (long)blockIdx.x * 64;
  for (int idx = tid; idx < 64 * 45; idx += 256) {           // float4 staging of y
    int i = idx / 45, k4 = idx - i * 45;
    float4 v = *(const float4*)(y + (Pbase + i) * 180 + k4 * 4);
    int k = k4 * 4;
    sA[i * 192 + k + 0] = (__bf16)v.x;
    sA[i * 192 + k + 1] = (__bf16)v.y;
    sA[i * 192 + k + 2] = (__bf16)v.z;
    sA[i * 192 + k + 3] = (__bf16)v.w;
  }
  for (int idx = tid; idx < 64 * 12; idx += 256) {
    int i = idx / 12, k = 180 + idx % 12;
    sA[i * 192 + k] = (__bf16)0.f;
  }
  const int wave = tid >> 5, lane = tid & 31;
  v8f acc[6] = {{}, {}, {}, {}, {}, {}};
  for (int kc = 0; kc < 6; ++kc) {
    __syncthreads();
    for (int idx = tid; idx < 192 * 32; idx += 256) {
      int o = idx >> 5, kk = idx & 31;
      int c = kc * 32 + kk;
      float v = (o < 180 && c < 180) ? pw[c * 180 + o] : 0.f;
      sBT[o * 32 + kk] = (__bf16)v;
    }
    __syncthreads();
#pragma unroll
    for (int s = 0; s < 6; ++s) {
      int t = wave + 8 * s;
      int msub = t & 3, nt = t >> 2;
      v16bf a = frag_a_bf16(sA, 192, msub * 16, kc * 32);
      v16bf bb = frag_b_bf16(sBT, 32, nt * 16, 0);
      acc[s] = WMMA_BF16(a, bb, acc[s]);
    }
  }
#pragma unroll
  for (int s = 0; s < 6; ++s) {
    int t = wave + 8 * s;
    int msub = t & 3, nt = t >> 2;
    int n = nt * 16 + (lane & 15);
    float bias = (n < 180) ? pb[n] : 0.f;
#pragma unroll
    for (int r = 0; r < 8; ++r) {
      int m = msub * 16 + r + ((lane >> 4) << 3);
      sOut[m * 192 + n] = acc[s][r] + bias;
    }
  }
  __syncthreads();
  {
    int r = tid >> 2, q = tid & 3;
    float ssum = 0.f;
    for (int n = q * 48; n < q * 48 + 48; ++n) {
      float v = sOut[r * 192 + n];
      ssum += v * v;
    }
    sPart[r * 4 + q] = ssum;
  }
  __syncthreads();
  if (tid < 64) {
    float ssum = sPart[tid * 4] + sPart[tid * 4 + 1] + sPart[tid * 4 + 2] + sPart[tid * 4 + 3];
    sScale[tid] = rsqrtf(ssum * (1.f / 180.f) + 1.1920929e-07f);
  }
  __syncthreads();
  const long b = Pbase >> 16;
  const long hw0 = Pbase & 65535;
  for (int idx = tid; idx < 180 * 64; idx += 256) {
    int c = idx >> 6, i = idx & 63;
    long addr = ((b * 180 + c) << 16) + hw0 + i;
    out[addr] = x[addr] + sOut[i * 192 + c] * sScale[i] * nw[c];
  }
}

// ---------------------------------------------------------------------------
// Launch
// ---------------------------------------------------------------------------
extern "C" void kernel_launch(void* const* d_in, const int* in_sizes, int n_in,
                              void* d_out, int out_size, void* d_ws, size_t ws_size,
                              hipStream_t stream) {
  (void)in_sizes; (void)n_in; (void)out_size; (void)ws_size;
  const float* x      = (const float*)d_in[0];
  const float* dfe_w1 = (const float*)d_in[1];
  const float* dfe_b1 = (const float*)d_in[2];
  const float* dfe_w2 = (const float*)d_in[3];
  const float* dfe_b2 = (const float*)d_in[4];
  const float* dfe_w3 = (const float*)d_in[5];
  const float* dfe_b3 = (const float*)d_in[6];
  const float* dfe_lw = (const float*)d_in[7];
  const float* dfe_lb = (const float*)d_in[8];
  const float* sl_w   = (const float*)d_in[9];
  const float* sl_b   = (const float*)d_in[10];
  const float* pos_pw = (const float*)d_in[11];
  const float* pos_pb = (const float*)d_in[12];
  const float* ln1w   = (const float*)d_in[13];
  const float* ln1b   = (const float*)d_in[14];
  const float* l1w    = (const float*)d_in[15];
  const float* l1b    = (const float*)d_in[16];
  const float* ln2w   = (const float*)d_in[17];
  const float* ln2b   = (const float*)d_in[18];
  const float* l2w    = (const float*)d_in[19];
  const float* l2b    = (const float*)d_in[20];
  const float* ln3w   = (const float*)d_in[21];
  const float* ln3b   = (const float*)d_in[22];
  const float* l3w    = (const float*)d_in[23];
  const float* l3b    = (const float*)d_in[24];
  const float* proj_w = (const float*)d_in[25];
  const float* proj_b = (const float*)d_in[26];
  const float* norm_w = (const float*)d_in[27];
  float* out = (float*)d_out;

  // Workspace carve (bytes): t1/t2 bf16, rpb/qv/y f32
  char* ws = (char*)d_ws;
  __bf16* t1 = (__bf16*)ws;                                   // 18,874,368
  __bf16* t2 = (__bf16*)(ws + 18874368);                      // 18,874,368
  float*  rpb = (float*)(ws + 37748736);                      // 393,216
  float*  qv  = (float*)(ws + 38141952);                      // 188,743,680
  float*  yb  = (float*)(ws + 226885632);                     // 188,743,680

  k_dfe1<<<4096, 256, 0, stream>>>(x, dfe_w1, dfe_b1, t1);
  k_dfe2<<<4096, 256, 0, stream>>>(t1, dfe_w2, dfe_b2, t2);
  k_gate<<<4096, 256, 0, stream>>>(x, t2, dfe_w3, dfe_b3, dfe_lw, dfe_lb, qv);
  k_pos<<<1, 256, 0, stream>>>(pos_pw, pos_pb, ln1w, ln1b, l1w, l1b,
                               ln2w, ln2b, l2w, l2b, ln3w, ln3b, l3w, l3b, rpb);
  k_attn_sp<<<1024, 256, 0, stream>>>(qv, sl_w, sl_b, rpb, yb);
  k_attn_ch<<<1024, 256, 0, stream>>>(qv, yb);
  k_proj<<<4096, 256, 0, stream>>>(x, yb, proj_w, proj_b, norm_w, out);
}
